// IterRefineODE_48421461295721
// MI455X (gfx1250) — matrix operand order repaired and checked by reading, not compiled, not run on hardware
//
#include <hip/hip_runtime.h>
#include <hip/hip_bf16.h>
#include <cstdint>
#include <cstddef>

typedef _Float16 f16;
typedef __attribute__((ext_vector_type(16))) _Float16 v16h;
typedef __attribute__((ext_vector_type(8)))  _Float16 v8h;
typedef __attribute__((ext_vector_type(8)))  float    v8f;

static constexpr int BB = 2, HH = 48, WWY = 40, DDX = 48;
static constexpr int SS  = HH * WWY * DDX;        // 92160
static constexpr int NSP = BB * SS;               // 184320
static constexpr int CTOT = 368;                  // fused dense-net channel buffer
static constexpr int NY = 4;                      // y-rows per conv block
static constexpr int NTILE4 = BB * HH * (WWY / NY) * (DDX / 16); // 2880 conv blocks

// ---------------------------------------------------------------- utilities

__device__ inline void tri_setup(float px, float py, float pz,
                                 int idx[8], float wgt[8], bool val[8]) {
  float fx = floorf(px), fy = floorf(py), fz = floorf(pz);
  float wx = px - fx, wy = py - fy, wz = pz - fz;
  int x0 = (int)fx, y0 = (int)fy, z0 = (int)fz;
#pragma unroll
  for (int c = 0; c < 8; ++c) {
    int ix = x0 + ((c >> 2) & 1);
    int iy = y0 + ((c >> 1) & 1);
    int iz = z0 + (c & 1);
    bool v = (ix >= 0 && ix < HH && iy >= 0 && iy < WWY && iz >= 0 && iz < DDX);
    val[c] = v;
    int ixc = min(max(ix, 0), HH - 1);
    int iyc = min(max(iy, 0), WWY - 1);
    int izc = min(max(iz, 0), DDX - 1);
    idx[c] = (ixc * WWY + iyc) * DDX + izc;
    float cwx = ((c >> 2) & 1) ? wx : 1.f - wx;
    float cwy = ((c >> 1) & 1) ? wy : 1.f - wy;
    float cwz = (c & 1) ? wz : 1.f - wz;
    wgt[c] = cwx * cwy * cwz;
  }
}

// ---------------------------------------------------------------- K0: zero out + stats accumulators

__global__ void zero_kernel(float* __restrict__ out, size_t n, float* __restrict__ accum) {
  size_t i = (size_t)blockIdx.x * blockDim.x + threadIdx.x;
  size_t stride = (size_t)gridDim.x * blockDim.x;
  for (size_t p = i; p < n; p += stride) out[p] = 0.f;
  if (i < 16) accum[i] = 0.f;
}

// ---------------------------------------------------------------- weight f32 -> f16
// K enumeration: blk(32 ci) -> tap(27) -> ciL(32); ciL zero-padded past CI.
// Within each 32-chunk, positions are pre-permuted into the HW A-fragment
// register order: stored p holds logical slot (p<8:p | p<16:p+8 | p<24:p-8 | p).

__global__ void wconv_kernel(const float* __restrict__ w, f16* __restrict__ dst,
                             int CO, int CI, int Kpad) {
  int idx = blockIdx.x * 256 + threadIdx.x;
  if (idx >= CO * Kpad) return;
  int co = idx / Kpad, p = idx - co * Kpad;
  int blk = p / 864;
  int q = p - blk * 864;
  int tap = q >> 5;
  int within = q & 31;
  int ciL = (within < 8) ? within
          : (within < 16) ? within + 8
          : (within < 24) ? within - 8
          : within;
  int ci = blk * 32 + ciL;
  float v = (ci < CI) ? w[(co * CI + ci) * 27 + tap] : 0.f;
  dst[idx] = (f16)v;
}

// ---------------------------------------------------------------- K1: warp(feature_x, disp) + stats + f16 packing

__global__ void warp_stats_kernel(const float* __restrict__ x, float* __restrict__ warped,
                                  f16* __restrict__ xin, float* __restrict__ accum) {
  int p = blockIdx.x * 256 + threadIdx.x;           // NSP threads exactly
  int b = p / SS, sp = p - b * SS;
  int z = sp / (WWY * DDX);
  int r = sp - z * WWY * DDX;
  int y = r / DDX, d = r - y * DDX;
  const float* xb = x + (size_t)b * 100 * SS;

  float d0 = xb[0 * SS + sp], d1 = xb[1 * SS + sp], d2 = xb[2 * SS + sp];
  int cidx[8]; float cw[8]; bool cv[8];
  tri_setup((float)z + d0, (float)y + d1, (float)d + d2, cidx, cw, cv);

  float sw = 0.f, sw2 = 0.f, sy = 0.f, sy2 = 0.f;
  for (int c = 0; c < 32; ++c) {
    const float* fx = xb + (size_t)(3 + c) * SS;
    float acc = 0.f;
#pragma unroll
    for (int k = 0; k < 8; ++k) acc += cv[k] ? cw[k] * fx[cidx[k]] : 0.f;
    warped[(size_t)(b * 32 + c) * SS + sp] = acc;
    xin[(size_t)(b * CTOT + 157 + c) * SS + sp] = (f16)acc;
    sw += acc; sw2 += acc * acc;
    float fy = xb[(size_t)(35 + c) * SS + sp];
    xin[(size_t)(b * CTOT + 125 + c) * SS + sp] = (f16)fy;
    sy += fy; sy2 += fy * fy;
    float cx = xb[(size_t)(67 + c) * SS + sp];
    xin[(size_t)(b * CTOT + 189 + c) * SS + sp] = (f16)cx;
  }
  xin[(size_t)(b * CTOT + 221) * SS + sp] = (f16)d0;
  xin[(size_t)(b * CTOT + 222) * SS + sp] = (f16)d1;
  xin[(size_t)(b * CTOT + 223) * SS + sp] = (f16)d2;
  float tval = xb[(size_t)99 * SS + sp];

  __shared__ float sred[5];
  if (threadIdx.x < 5) sred[threadIdx.x] = 0.f;
  __syncthreads();
  atomicAdd(&sred[0], sw);  atomicAdd(&sred[1], sw2);
  atomicAdd(&sred[2], sy);  atomicAdd(&sred[3], sy2);
  atomicAdd(&sred[4], tval);
  __syncthreads();
  if (threadIdx.x == 0) {
    atomicAdd(&accum[b * 4 + 0], sred[0]);
    atomicAdd(&accum[b * 4 + 1], sred[1]);
    atomicAdd(&accum[b * 4 + 2], sred[2]);
    atomicAdd(&accum[b * 4 + 3], sred[3]);
    atomicAdd(&accum[8], sred[4]);
  }
}

// ---------------------------------------------------------------- K2: finalize stats + 1/(t_end - t)

__global__ void finalize_kernel(const float* __restrict__ accum, const float* __restrict__ t,
                                float* __restrict__ fin) {
  if (blockIdx.x != 0 || threadIdx.x != 0) return;
  const float nf = 32.f * (float)SS;
  for (int b = 0; b < BB; ++b) {
    float s1 = accum[b * 4 + 0], s1q = accum[b * 4 + 1];
    float s2 = accum[b * 4 + 2], s2q = accum[b * 4 + 3];
    float m1 = s1 / nf, m2 = s2 / nf;
    float v1 = (s1q - s1 * s1 / nf) / (nf - 1.f);
    float v2 = (s2q - s2 * s2 / nf) / (nf - 1.f);
    fin[b] = 0.5f * (m1 + m2);
    fin[2 + b] = 1.f / (0.5f * (sqrtf(v1) + sqrtf(v2)));
  }
  float t_end = accum[8] / (float)NSP;
  fin[4] = 1.f / (t_end - t[0]);
}

// ---------------------------------------------------------------- K3: 125-shift cost volume (+leaky 0.05) -> f16 ch 0..124

__global__ void cost_kernel(const float* __restrict__ x, const float* __restrict__ warped,
                            const float* __restrict__ fin, f16* __restrict__ xin) {
  int p = blockIdx.x * 256 + threadIdx.x;
  int b = p / SS, sp = p - b * SS;
  int z = sp / (WWY * DDX);
  int r = sp - z * WWY * DDX;
  int y = r / DDX, d = r - y * DDX;
  float m = fin[b], invs = fin[2 + b];
  const float* xb = x + (size_t)b * 100 * SS;

  float fy[32];
#pragma unroll
  for (int c = 0; c < 32; ++c) fy[c] = (xb[(size_t)(35 + c) * SS + sp] - m) * invs;

  const float* wb = warped + (size_t)b * 32 * SS;
  int sidx = 0;
  for (int i = -2; i <= 2; ++i)
    for (int j = -2; j <= 2; ++j)
      for (int k = -2; k <= 2; ++k, ++sidx) {
        int zz = z + i, yy = y + j, dd = d + k;
        float cst = 0.f;
        if (zz >= 0 && zz < HH && yy >= 0 && yy < WWY && dd >= 0 && dd < DDX) {
          int sps = (zz * WWY + yy) * DDX + dd;
          float a = 0.f;
#pragma unroll
          for (int c = 0; c < 32; ++c) a += fy[c] * (wb[(size_t)c * SS + sps] - m);
          cst = a * invs * (1.f / 32.f);
        }
        cst = cst >= 0.f ? cst : 0.05f * cst;
        xin[(size_t)(b * CTOT + sidx) * SS + sp] = (f16)cst;
      }
}

// ---------------------------------------------------------------- implicit-GEMM 3x3x3 conv via WMMA f16->f32
// Block: (b, z, y0..y0+3, d0..d0+15).  K-chunk == one (ci-block, tap): per chunk
// the A fragment is one contiguous 32B load (pre-permuted weights) and each B
// fragment is one contiguous 32B LDS load.  NWC waves split output channels,
// NWN waves split the 4 y-tiles; each A fragment is reused for TPW WMMAs.
// Fragment layouts per ISA 7.12.2 (16-bit A 16x32; B: lanes0-15 K0..7,16..23, lanes16-31 K8..15,24..31).

template <int CI, int CO, int NWC, int NWN>
__global__ __launch_bounds__(32 * NWC * NWN)
void conv_wmma_kernel(const f16* __restrict__ xin, const f16* __restrict__ Aw,
                      const float* __restrict__ bias, f16* __restrict__ outp,
                      int outCtot, int outChBase) {
  constexpr int TPW   = NY / NWN;          // y-tiles per wave
  constexpr int NBLK  = (CI + 31) / 32;
  constexpr int KPAD  = NBLK * 864;
  constexpr int ROWS  = 3 * (NY + 2);      // 18 (dz,dyy) rows
  constexpr int CELLS = ROWS * 18;         // 324 (row, dd) cells
  constexpr int CP    = 40;                // cell stride in f16 (bank-spread pad)
  constexpr int NT    = 32 * NWC * NWN;
  __shared__ __align__(32) f16 lds[CELLS * CP];   // 25920 B

  int tid = threadIdx.x, lane = tid & 31, wave = tid >> 5;
  int coTile = wave % NWC;
  int nGrp   = wave / NWC;
  int n = lane & 15, half = lane >> 4;

  int bx = blockIdx.x;
  int dt = bx % 3; int rem = bx / 3;
  int y4 = rem % (WWY / NY); rem /= (WWY / NY);
  int z = rem % HH; int b = rem / HH;
  int d0 = dt * 16, y0 = y4 * NY;

  const f16* arow = Aw + (size_t)(coTile * 16 + n) * KPAD + half * 16;
  v8f acc[TPW] = {};

  for (int blk = 0; blk < NBLK; ++blk) {
    int c0 = blk * 32;
    __syncthreads();
    // stage 32 input channels x 18 (dz,dyy) rows x 18 d positions
    for (int idx = tid; idx < CELLS * 32; idx += NT) {
      int ciL = idx & 31;
      int cell = idx >> 5;                 // 0..323
      int row = cell / 18, dd = cell - row * 18;
      int dz = row / (NY + 2), dyy = row - dz * (NY + 2);
      int zz = z + dz - 1, yy = y0 + dyy - 1, dp = d0 + dd - 1;
      int ch = c0 + ciL;
      f16 v = (f16)0.f;
      if (ch < CI && zz >= 0 && zz < HH && yy >= 0 && yy < WWY && dp >= 0 && dp < DDX)
        v = xin[(size_t)(b * CTOT + ch) * SS + (zz * WWY + yy) * DDX + dp];
      lds[cell * CP + (((ciL + 8) & 16) ? (ciL ^ 24) : ciL)] = v;
    }
    __syncthreads();

    const f16* ab = arow + blk * 864;
#pragma unroll
    for (int dz = 0; dz < 3; ++dz)
#pragma unroll
      for (int dy = 0; dy < 3; ++dy)
#pragma unroll
        for (int dx = 0; dx < 3; ++dx) {
          int tap = (dz * 3 + dy) * 3 + dx;
          v16h a = *(const v16h*)(ab + tap * 32);       // one 32B A fragment
#pragma unroll
          for (int j = 0; j < TPW; ++j) {
            int yt = nGrp * TPW + j;
            int row = dz * (NY + 2) + dy + yt;
            const f16* bp = &lds[(row * 18 + n + dx) * CP + half * 16];
            v16h bf = *(const v16h*)bp;                 // one 32B B fragment
            acc[j] = __builtin_amdgcn_wmma_f32_16x16x32_f16(
                false, a, false, bf, (short)0, acc[j], false, false);
          }
        }
  }

  // epilogue: bias + leaky(0.02) + f16 store (D: lane<16 M=r, lane>=16 M=8+r)
#pragma unroll
  for (int j = 0; j < TPW; ++j) {
    int yt = nGrp * TPW + j;
    int sp = (z * WWY + (y0 + yt)) * DDX + d0 + n;
#pragma unroll
    for (int r2 = 0; r2 < 8; ++r2) {
      int co = coTile * 16 + half * 8 + r2;
      float v = acc[j][r2] + bias[co];
      v = v >= 0.f ? v : 0.02f * v;
      outp[(size_t)(b * outCtot + outChBase + co) * SS + sp] = (f16)v;
    }
  }
}

// ---------------------------------------------------------------- final 16->3 conv (tiny, direct)

__global__ void conv4_kernel(const f16* __restrict__ out3, const float* __restrict__ w4,
                             const float* __restrict__ b4, float* __restrict__ vel) {
  int p = blockIdx.x * 256 + threadIdx.x;
  int b = p / SS, sp = p - b * SS;
  int z = sp / (WWY * DDX);
  int r = sp - z * WWY * DDX;
  int y = r / DDX, d = r - y * DDX;
  float a0 = b4[0], a1 = b4[1], a2 = b4[2];
  for (int dz = -1; dz <= 1; ++dz) {
    int zz = z + dz; if (zz < 0 || zz >= HH) continue;
    for (int dy = -1; dy <= 1; ++dy) {
      int yy = y + dy; if (yy < 0 || yy >= WWY) continue;
      for (int dx = -1; dx <= 1; ++dx) {
        int dd = d + dx; if (dd < 0 || dd >= DDX) continue;
        int tap = (dz + 1) * 9 + (dy + 1) * 3 + (dx + 1);
        int spn = (zz * WWY + yy) * DDX + dd;
#pragma unroll
        for (int ci = 0; ci < 16; ++ci) {
          float v = (float)out3[(size_t)(b * 16 + ci) * SS + spn];
          a0 += w4[(0 * 16 + ci) * 27 + tap] * v;
          a1 += w4[(1 * 16 + ci) * 27 + tap] * v;
          a2 += w4[(2 * 16 + ci) * 27 + tap] * v;
        }
      }
    }
  }
  vel[(size_t)(b * 3 + 0) * SS + sp] = a0;
  vel[(size_t)(b * 3 + 1) * SS + sp] = a1;
  vel[(size_t)(b * 3 + 2) * SS + sp] = a2;
}

// ---------------------------------------------------------------- epilogue: warp(disp, vel) + vel - disp, scaled

__global__ void final_kernel(const float* __restrict__ x, const float* __restrict__ vel,
                             const float* __restrict__ fin, float* __restrict__ out) {
  int p = blockIdx.x * 256 + threadIdx.x;
  int b = p / SS, sp = p - b * SS;
  int z = sp / (WWY * DDX);
  int r = sp - z * WWY * DDX;
  int y = r / DDX, d = r - y * DDX;
  const float* xb = x + (size_t)b * 100 * SS;
  float v0 = vel[(size_t)(b * 3 + 0) * SS + sp];
  float v1 = vel[(size_t)(b * 3 + 1) * SS + sp];
  float v2 = vel[(size_t)(b * 3 + 2) * SS + sp];
  int cidx[8]; float cw[8]; bool cv[8];
  tri_setup((float)z + v0, (float)y + v1, (float)d + v2, cidx, cw, cv);
  float invdt = fin[4];
  float* ob = out + (size_t)b * 100 * SS;
#pragma unroll
  for (int c = 0; c < 3; ++c) {
    const float* dv = xb + (size_t)c * SS;
    float wv = 0.f;
#pragma unroll
    for (int k = 0; k < 8; ++k) wv += cv[k] ? cw[k] * dv[cidx[k]] : 0.f;
    float vc = (c == 0 ? v0 : (c == 1 ? v1 : v2));
    ob[(size_t)c * SS + sp] = (wv + vc - dv[sp]) * invdt;
  }
}

// ---------------------------------------------------------------- host launcher

extern "C" void kernel_launch(void* const* d_in, const int* in_sizes, int n_in,
                              void* d_out, int out_size, void* d_ws, size_t ws_size,
                              hipStream_t stream) {
  const float* t  = (const float*)d_in[0];
  const float* x  = (const float*)d_in[1];
  const float* w0 = (const float*)d_in[2];  const float* b0 = (const float*)d_in[3];
  const float* w1 = (const float*)d_in[4];  const float* b1 = (const float*)d_in[5];
  const float* w2 = (const float*)d_in[6];  const float* b2 = (const float*)d_in[7];
  const float* w3 = (const float*)d_in[8];  const float* b3 = (const float*)d_in[9];
  const float* w4 = (const float*)d_in[10]; const float* b4 = (const float*)d_in[11];

  char* ws = (char*)d_ws;
  size_t off = 0;
  auto alloc = [&](size_t bytes) -> void* {
    void* p = ws + off;
    off = (off + bytes + 255) & ~(size_t)255;
    return p;
  };

  // KPADs: ceil(CI/32) * 864
  constexpr int KP0 = 7 * 864;    // 6048
  constexpr int KP1 = 9 * 864;    // 7776
  constexpr int KP2 = 11 * 864;   // 9504
  constexpr int KP3 = 12 * 864;   // 10368

  f16*   xin   = (f16*)alloc((size_t)CTOT * NSP * sizeof(f16));
  float* wrp   = (float*)alloc((size_t)32 * NSP * sizeof(float));
  f16*   out3  = (f16*)alloc((size_t)16 * NSP * sizeof(f16));
  float* vel   = (float*)alloc((size_t)3 * NSP * sizeof(float));
  f16*   aw0   = (f16*)alloc((size_t)64 * KP0 * sizeof(f16));
  f16*   aw1   = (f16*)alloc((size_t)48 * KP1 * sizeof(f16));
  f16*   aw2   = (f16*)alloc((size_t)32 * KP2 * sizeof(f16));
  f16*   aw3   = (f16*)alloc((size_t)16 * KP3 * sizeof(f16));
  float* accum = (float*)alloc(64);
  float* fin   = (float*)alloc(64);

  // 0) zero output (channels 3..99 stay zero) + stats accumulators
  zero_kernel<<<4096, 256, 0, stream>>>((float*)d_out, (size_t)BB * 100 * SS, accum);

  // 1) weights -> f16, blocked/permuted K layout
  wconv_kernel<<<(64 * KP0 + 255) / 256, 256, 0, stream>>>(w0, aw0, 64, 224, KP0);
  wconv_kernel<<<(48 * KP1 + 255) / 256, 256, 0, stream>>>(w1, aw1, 48, 288, KP1);
  wconv_kernel<<<(32 * KP2 + 255) / 256, 256, 0, stream>>>(w2, aw2, 32, 336, KP2);
  wconv_kernel<<<(16 * KP3 + 255) / 256, 256, 0, stream>>>(w3, aw3, 16, 368, KP3);

  // 2) warp + stats + f16 packing of f_y / warped / ctx / disp
  warp_stats_kernel<<<NSP / 256, 256, 0, stream>>>(x, wrp, xin, accum);

  // 3) normalization constants + 1/(t_end - t)
  finalize_kernel<<<1, 1, 0, stream>>>(accum, t, fin);

  // 4) cost volume -> xin channels 0..124
  cost_kernel<<<NSP / 256, 256, 0, stream>>>(x, wrp, fin, xin);

  // 5) dense-net conv layers (implicit GEMM, WMMA f16)
  conv_wmma_kernel<224, 64, 4, 1><<<NTILE4, 128, 0, stream>>>(xin, aw0, b0, xin, CTOT, 224);
  conv_wmma_kernel<288, 48, 3, 1><<<NTILE4,  96, 0, stream>>>(xin, aw1, b1, xin, CTOT, 288);
  conv_wmma_kernel<336, 32, 2, 1><<<NTILE4,  64, 0, stream>>>(xin, aw2, b2, xin, CTOT, 336);
  conv_wmma_kernel<368, 16, 1, 2><<<NTILE4,  64, 0, stream>>>(xin, aw3, b3, out3,  16,   0);

  // 6) 16->3 conv -> raw velocity
  conv4_kernel<<<NSP / 256, 256, 0, stream>>>(out3, w4, b4, vel);

  // 7) warp-compose + scale -> output channels 0..2
  final_kernel<<<NSP / 256, 256, 0, stream>>>(x, vel, fin, (float*)d_out);

  (void)in_sizes; (void)n_in; (void)out_size; (void)ws_size;
}
